// ESAValidationModel_54365696032882
// MI455X (gfx1250) — compile-verified, matrix operand order checked by reading, not run on hardware
//
#include <hip/hip_runtime.h>
#include <hip/hip_bf16.h>
#include <math.h>

typedef _Float16 h16;
typedef __attribute__((ext_vector_type(16))) _Float16 v16h;
typedef __attribute__((ext_vector_type(8)))  _Float16 v8h;
typedef __attribute__((ext_vector_type(8)))  float    v8f;

#define WMMA16(a,b,c) __builtin_amdgcn_wmma_f32_16x16x32_f16(false,(a),false,(b),(short)0,(c),false,false)

#define NTOK  43808
#define NPB   10952
#define PADN  11008
#define PP    1369
#define DN    256
#define KSLOT 33

__device__ __forceinline__ float wave_sum(float v){
  #pragma unroll
  for (int m = 16; m > 0; m >>= 1) v += __shfl_xor(v, m, 32);
  return v;
}

__device__ __forceinline__ float block_sum(float v, float* red){
  int tid = threadIdx.x;
  red[tid] = v; __syncthreads();
  #pragma unroll
  for (int s = 128; s > 0; s >>= 1){
    if (tid < s) red[tid] += red[tid + s];
    __syncthreads();
  }
  float r = red[0]; __syncthreads();
  return r;
}

__device__ __forceinline__ float gelu_exact(float x){
  return 0.5f * x * (1.0f + erff(x * 0.70710678118654752f));
}

__device__ __forceinline__ float sigm(float x){ return 1.0f / (1.0f + expf(-x)); }

// Load a 16x32 f16 fragment (A layout; B uses same layout with row==N of the
// transposed-B storage). `row` must already include the lane's row index.
__device__ __forceinline__ v16h load_frag_rows(const h16* base, int ld, int row, int k0){
  int hi = (threadIdx.x >> 4) & 1;
  const h16* p = base + (size_t)row * ld + k0 + hi * 8;
  union { v16h v; v8h h[2]; } u;
  u.h[0] = *(const v8h*)p;
  u.h[1] = *(const v8h*)(p + 16);
  return u.v;
}

__device__ __forceinline__ v16h load_frag(const h16* base, int ld, int row0, int k0){
  int l = threadIdx.x & 15;
  return load_frag_rows(base, ld, row0 + l, k0);
}

// A-fragment straight from fp32 features with fused LayerNorm + f16 cast.
__device__ __forceinline__ v16h load_fragA_ln(const float* feat, const float2* stats,
                                              const float* g1, const float* b1,
                                              int row0, int k0){
  int lane = threadIdx.x & 31; int l = lane & 15; int hi = lane >> 4;
  int row = row0 + l;
  float2 st = stats[row];
  const float* p = feat + (size_t)row * 2048 + k0 + hi * 8;
  __builtin_prefetch(p + 2048, 0, 1);
  union { v16h v; _Float16 e[16]; } o;
  #pragma unroll
  for (int i = 0; i < 8; ++i){
    int kk = k0 + hi * 8 + i;
    float x0 = p[i];
    float x1 = p[16 + i];
    o.e[i]     = (_Float16)((x0 - st.x) * st.y * g1[kk]      + b1[kk]);
    o.e[8 + i] = (_Float16)((x1 - st.x) * st.y * g1[kk + 16] + b1[kk + 16]);
  }
  return o.v;
}

// ---------------------------------------------------------------- prep kernels
__global__ void k_transpose_f16(h16* dst, const float* src, int A, int B){
  int idx = blockIdx.x * 256 + threadIdx.x;
  if (idx >= A * B) return;
  int j = idx / A; int i = idx % A;               // dst[j][i] = src[i][j]
  dst[(size_t)j * A + i] = (h16)src[(size_t)i * B + j];
}

__global__ void k_transpose_f32(float* dst, const float* src, int A, int B){
  int idx = blockIdx.x * 256 + threadIdx.x;
  if (idx >= A * B) return;
  int j = idx / A; int i = idx % A;
  dst[(size_t)j * A + i] = src[(size_t)i * B + j];
}

__global__ void k_zero_vt_tail(h16* vT){
  int idx = blockIdx.x * 256 + threadIdx.x;
  const int total = 4 * 256 * (PADN - NPB);
  if (idx >= total) return;
  int per_b = 256 * (PADN - NPB);
  int b = idx / per_b; int r = idx % per_b;
  int n = r / (PADN - NPB); int t = r % (PADN - NPB);
  vT[((size_t)(b * 256 + n)) * PADN + NPB + t] = (h16)0.0f;
}

__global__ void k_init_slots(float* out_slots, const float* slot_embed){
  int idx = blockIdx.x * 256 + threadIdx.x;
  if (idx >= 4 * KSLOT * DN) return;
  out_slots[idx] = slot_embed[idx % (KSLOT * DN)];
}

// per-token LN statistics over IN_DIM=2048
__global__ __launch_bounds__(256) void k_ln_stats(const float* feat, float2* stats){
  __shared__ float red[256];
  int tok = blockIdx.x; int tid = threadIdx.x;
  const float* p = feat + (size_t)tok * 2048;
  float s = 0.f, s2 = 0.f;
  #pragma unroll
  for (int i = 0; i < 8; ++i){ float x = p[tid + 256 * i]; s += x; s2 += x * x; }
  s  = block_sum(s,  red);
  s2 = block_sum(s2, red);
  if (tid == 0){
    float m = s * (1.0f / 2048.0f);
    float v = s2 * (1.0f / 2048.0f) - m * m;
    stats[tok] = make_float2(m, rsqrtf(v + 1e-5f));
  }
}

// ------------------------------------------------- projector GEMM (WMMA, f16)
// [32 tok/block x 2048] @ WPt -> 256, fused bias+GELU+LN2+embeds+LN(ni) -> inp f16
__global__ __launch_bounds__(256) void k_proj_gemm(
    const float* feat, const float2* stats, const float* g1, const float* b1,
    const h16* WPt, const float* pbias, const float* g2, const float* b2,
    const float* nig, const float* nib, const float* pos, const float* frame,
    h16* inp)
{
  __shared__ float ldsC[32][256];
  int t0 = blockIdx.x * 32;
  int w = threadIdx.x >> 5; int lane = threadIdx.x & 31;
  int mt = w & 1; int ng = w >> 1;                 // wave: 16 rows x 64 cols
  v8f c[4] = {};
  for (int kt = 0; kt < 64; ++kt){
    int k0 = kt * 32;
    v16h a = load_fragA_ln(feat, stats, g1, b1, t0 + mt * 16, k0);
    #pragma unroll
    for (int j = 0; j < 4; ++j){
      v16h bf = load_frag(WPt, 2048, (ng * 4 + j) * 16, k0);
      c[j] = WMMA16(a, bf, c[j]);
    }
  }
  int l = lane & 15; int hi = lane >> 4;
  #pragma unroll
  for (int j = 0; j < 4; ++j){
    int ncol = (ng * 4 + j) * 16 + l;
    #pragma unroll
    for (int r = 0; r < 8; ++r){
      int row = mt * 16 + (hi ? 8 + r : r);
      float x = c[j][r] + pbias[ncol];
      ldsC[row][ncol] = gelu_exact(x);
    }
  }
  __syncthreads();
  // epilogue: each wave owns 4 rows; per row: LN2 -> +pos+frame -> LN(ni) -> f16
  for (int rr = w * 4; rr < w * 4 + 4; ++rr){
    int tok = t0 + rr;
    int rem = tok % NPB; int sfr = rem / PP; int pp = rem % PP;
    float v0[8]; float s1 = 0.f;
    #pragma unroll
    for (int q = 0; q < 8; ++q){ v0[q] = ldsC[rr][lane + 32 * q]; s1 += v0[q]; }
    s1 = wave_sum(s1); float m = s1 * (1.0f / 256.0f);
    float s2 = 0.f;
    #pragma unroll
    for (int q = 0; q < 8; ++q){ float d = v0[q] - m; s2 += d * d; }
    s2 = wave_sum(s2); float rs = rsqrtf(s2 * (1.0f / 256.0f) + 1e-5f);
    float e[8]; float t1 = 0.f;
    #pragma unroll
    for (int q = 0; q < 8; ++q){
      int cc = lane + 32 * q;
      e[q] = (v0[q] - m) * rs * g2[cc] + b2[cc]
           + pos[(size_t)pp * 256 + cc] + frame[(size_t)sfr * 256 + cc];
      t1 += e[q];
    }
    t1 = wave_sum(t1); float m2 = t1 * (1.0f / 256.0f);
    float t2 = 0.f;
    #pragma unroll
    for (int q = 0; q < 8; ++q){ float d = e[q] - m2; t2 += d * d; }
    t2 = wave_sum(t2); float rs2 = rsqrtf(t2 * (1.0f / 256.0f) + 1e-5f);
    #pragma unroll
    for (int q = 0; q < 8; ++q){
      int cc = lane + 32 * q;
      inp[(size_t)tok * 256 + cc] = (h16)((e[q] - m2) * rs2 * nig[cc] + nib[cc]);
    }
  }
}

// ----------------------------------------------- K/V GEMM (WMMA), N=512 fused
__global__ __launch_bounds__(256) void k_kv_gemm(
    const h16* inp, const h16* Wkvt, h16* k_h, h16* vT)
{
  __shared__ float ldsC[32][512];
  int t0 = blockIdx.x * 32;
  int w = threadIdx.x >> 5; int lane = threadIdx.x & 31;
  int mt = w & 1; int ng = w >> 1;                 // wave: 16 rows x 128 cols
  v8f c[8] = {};
  for (int kt = 0; kt < 8; ++kt){
    int k0 = kt * 32;
    v16h a = load_frag(inp, 256, t0 + mt * 16, k0);
    #pragma unroll
    for (int j = 0; j < 8; ++j){
      v16h bf = load_frag(Wkvt, 256, (ng * 8 + j) * 16, k0);
      c[j] = WMMA16(a, bf, c[j]);
    }
  }
  int l = lane & 15; int hi = lane >> 4;
  #pragma unroll
  for (int j = 0; j < 8; ++j){
    int ncol = (ng * 8 + j) * 16 + l;
    #pragma unroll
    for (int r = 0; r < 8; ++r){
      int row = mt * 16 + (hi ? 8 + r : r);
      ldsC[row][ncol] = c[j][r];
    }
  }
  __syncthreads();
  for (int rr = w * 4; rr < w * 4 + 4; ++rr){
    int tok = t0 + rr;
    int b = tok / NPB; int tl = tok % NPB;
    float kv[8]; float ss = 0.f;
    #pragma unroll
    for (int q = 0; q < 8; ++q){ kv[q] = ldsC[rr][lane + 32 * q]; ss += kv[q] * kv[q]; }
    ss = wave_sum(ss);
    float nrm = 1.0f / fmaxf(sqrtf(ss), 1e-12f);
    #pragma unroll
    for (int q = 0; q < 8; ++q)
      k_h[(size_t)tok * 256 + lane + 32 * q] = (h16)(kv[q] * nrm);
    #pragma unroll
    for (int q = 8; q < 16; ++q){
      int cc = lane + 32 * q;
      vT[((size_t)(b * 256 + (cc - 256))) * PADN + tl] = (h16)ldsC[rr][cc];
    }
  }
}

// ----------------------------------------------------------------- q kernel
__global__ __launch_bounds__(256) void k_slots_q(
    const float* slots, const float* nsg, const float* nsb,
    const float* wqT, h16* q_h)
{
  __shared__ float red[256];
  __shared__ float lnv[256];
  int bi = blockIdx.x; int b = bi / 48; int j = bi % 48; int tid = threadIdx.x;
  if (j >= KSLOT){ q_h[((size_t)(b * 48 + j)) * 256 + tid] = (h16)0.0f; return; }
  float s = slots[((size_t)(b * KSLOT + j)) * 256 + tid];
  float sm = block_sum(s, red) * (1.0f / 256.0f);
  float d  = s - sm;
  float vv = block_sum(d * d, red) * (1.0f / 256.0f);
  float rs = rsqrtf(vv + 1e-5f);
  lnv[tid] = d * rs * nsg[tid] + nsb[tid];
  __syncthreads();
  float q = 0.f;
  #pragma unroll 4
  for (int dd = 0; dd < 256; ++dd) q += lnv[dd] * wqT[(size_t)tid * 256 + dd];
  float ssq = block_sum(q * q, red);
  float nrm = 1.0f / fmaxf(sqrtf(ssq), 1e-12f);
  q_h[((size_t)(b * 48 + j)) * 256 + tid] = (h16)(q * nrm);
}

// --------------------------------------------- attention kernel (WMMA + softmax)
__global__ __launch_bounds__(256) void k_attn(
    const h16* q_h, const h16* k_h, float* attns_base, h16* attn_h,
    float* prev_bias, const float* logmean,
    const float* p_scale, const float* p_sink, const float* p_gate, int t)
{
  __shared__ float ldsL[48][128];
  int b = blockIdx.y; int n0 = blockIdx.x * 128;
  int w = threadIdx.x >> 5; int lane = threadIdx.x & 31;
  int l = lane & 15; int hi = lane >> 4;
  v8f c[3] = {};
  const h16* qb = q_h + (size_t)b * 48 * 256;
  for (int kt = 0; kt < 8; ++kt){
    int k0 = kt * 32;
    int tloc = n0 + w * 16 + l; if (tloc > NPB - 1) tloc = NPB - 1;
    v16h bf = load_frag_rows(k_h, 256, b * NPB + tloc, k0);
    #pragma unroll
    for (int mt = 0; mt < 3; ++mt){
      v16h a = load_frag(qb, 256, mt * 16, k0);
      c[mt] = WMMA16(a, bf, c[mt]);
    }
  }
  #pragma unroll
  for (int mt = 0; mt < 3; ++mt){
    #pragma unroll
    for (int r = 0; r < 8; ++r){
      int row = mt * 16 + (hi ? 8 + r : r);
      ldsL[row][w * 16 + l] = c[mt][r];
    }
  }
  __syncthreads();
  int tid = threadIdx.x;
  if (tid < 128){
    int tloc = n0 + tid;
    float* attn_t = attns_base + (size_t)t * 4 * KSLOT * NPB;
    if (tloc < NPB){
      float scale = fminf(fmaxf(p_scale[0], 0.1f), 10.0f);
      float sinkb = p_sink[0];
      float gate  = sigm(p_gate[0]);
      float lg[KSLOT];
      #pragma unroll
      for (int j = 0; j < KSLOT; ++j)
        lg[j] = fminf(fmaxf(ldsL[j][tid] * scale, -15.0f), 15.0f);
      if (t > 0){
        const float* prevA = attns_base + (size_t)(t - 1) * 4 * KSLOT * NPB;
        float strength = 0.2f * (float)t;           // 0.6 * t / 3
        #pragma unroll
        for (int j = 0; j < KSLOT; ++j){
          size_t idx = (size_t)(b * KSLOT + j) * NPB + tloc;
          float nb = strength * (logf(prevA[idx] + 1e-8f) - logmean[b * KSLOT + j]);
          float mb = (t == 1) ? nb : (gate * nb + (1.0f - gate) * prev_bias[idx]);
          if (j < 2 || j >= KSLOT - 1) mb = 0.0f;   // global + sink rows exempt
          prev_bias[idx] = mb;
          lg[j] += mb;
        }
      }
      lg[KSLOT - 1] += sinkb;
      float mx = -1e30f;
      #pragma unroll
      for (int j = 0; j < KSLOT; ++j) mx = fmaxf(mx, lg[j]);
      float ssum = 0.f;
      #pragma unroll
      for (int j = 0; j < KSLOT; ++j){ lg[j] = expf(lg[j] - mx); ssum += lg[j]; }
      float inv = 1.0f / ssum;
      #pragma unroll
      for (int j = 0; j < KSLOT; ++j){
        float a = lg[j] * inv;
        attn_t[(size_t)(b * KSLOT + j) * NPB + tloc] = a;
        attn_h[(size_t)(b * 48 + j) * PADN + tloc] = (h16)a;
      }
      #pragma unroll
      for (int j = KSLOT; j < 48; ++j)
        attn_h[(size_t)(b * 48 + j) * PADN + tloc] = (h16)0.0f;
    } else {
      for (int j = 0; j < 48; ++j)
        attn_h[(size_t)(b * 48 + j) * PADN + tloc] = (h16)0.0f;
    }
  }
}

// -------------------------------- per-(b,slot) row-sum + log-mean over tokens
__global__ __launch_bounds__(256) void k_attn_reduce(
    const float* attn_t, float* rowsum, float* logmean)
{
  __shared__ float red[256];
  int i = blockIdx.x; int tid = threadIdx.x;
  const float* p = attn_t + (size_t)i * NPB;
  float s = 0.f, ls = 0.f;
  for (int n = tid; n < NPB; n += 256){
    float a = p[n];
    s += a; ls += logf(a + 1e-8f);
  }
  s  = block_sum(s,  red);
  ls = block_sum(ls, red);
  if (tid == 0){ rowsum[i] = s; logmean[i] = ls * (1.0f / (float)NPB); }
}

// ------------------------------------------------ updates = attn @ v (WMMA)
__global__ __launch_bounds__(256) void k_updates(
    const h16* attn_h, const h16* vT, const float* rowsum, float* updates)
{
  int b = blockIdx.y; int nh = blockIdx.x;
  int w = threadIdx.x >> 5; int lane = threadIdx.x & 31;
  int l = lane & 15; int hi = lane >> 4;
  const h16* ab = attn_h + (size_t)b * 48 * PADN;
  const h16* vb = vT + (size_t)b * 256 * PADN;
  v8f c[3] = {};
  for (int kt = 0; kt < PADN / 32; ++kt){
    int k0 = kt * 32;
    v16h bf = load_frag(vb, PADN, nh * 128 + w * 16, k0);
    #pragma unroll
    for (int mt = 0; mt < 3; ++mt){
      v16h a = load_frag(ab, PADN, mt * 16, k0);
      c[mt] = WMMA16(a, bf, c[mt]);
    }
  }
  int dcol = nh * 128 + w * 16 + l;
  #pragma unroll
  for (int mt = 0; mt < 3; ++mt){
    #pragma unroll
    for (int r = 0; r < 8; ++r){
      int j = mt * 16 + (hi ? 8 + r : r);
      if (j < KSLOT){
        float u = c[mt][r] / (rowsum[b * KSLOT + j] + 1e-8f);
        updates[((size_t)(b * KSLOT + j)) * 256 + dcol] = u;
      }
    }
  }
}

// --------------------------------------------------- GRU + MLP residual (VALU)
__global__ __launch_bounds__(256) void k_gru_mlp(
    const float* updates, float* slots,
    const float* wih, const float* whh, const float* bih, const float* bhh,
    const float* nmg, const float* nmb,
    const float* w1T, const float* b1, const float* w2T, const float* b2)
{
  __shared__ float u[256], hh[256], lnv[256], hid[512], red[256];
  int i = blockIdx.x; int tid = threadIdx.x;
  size_t row = (size_t)i * 256;
  u[tid]  = updates[row + tid];
  hh[tid] = slots[row + tid];
  __syncthreads();
  float ir = bih[tid], iz = bih[256 + tid], in2 = bih[512 + tid];
  float hr = bhh[tid], hz = bhh[256 + tid], hn  = bhh[512 + tid];
  #pragma unroll 4
  for (int d = 0; d < 256; ++d){
    float uv = u[d], hv = hh[d];
    ir  += wih[(size_t)tid * 256 + d] * uv;
    iz  += wih[(size_t)(256 + tid) * 256 + d] * uv;
    in2 += wih[(size_t)(512 + tid) * 256 + d] * uv;
    hr  += whh[(size_t)tid * 256 + d] * hv;
    hz  += whh[(size_t)(256 + tid) * 256 + d] * hv;
    hn  += whh[(size_t)(512 + tid) * 256 + d] * hv;
  }
  float r = sigm(ir + hr), z = sigm(iz + hz);
  float nn = tanhf(in2 + r * hn);
  float s = (1.0f - z) * nn + z * hh[tid];
  float sm = block_sum(s, red) * (1.0f / 256.0f);
  float d0 = s - sm;
  float vv = block_sum(d0 * d0, red) * (1.0f / 256.0f);
  lnv[tid] = d0 * rsqrtf(vv + 1e-5f) * nmg[tid] + nmb[tid];
  __syncthreads();
  #pragma unroll
  for (int half = 0; half < 2; ++half){
    int jj = tid + half * 256;
    float acc = b1[jj];
    #pragma unroll 4
    for (int d = 0; d < 256; ++d) acc += lnv[d] * w1T[(size_t)jj * 256 + d];
    hid[jj] = gelu_exact(acc);
  }
  __syncthreads();
  float o = s + b2[tid];
  #pragma unroll 4
  for (int j = 0; j < 512; ++j) o += hid[j] * w2T[(size_t)tid * 512 + j];
  slots[row + tid] = o;
}

// ============================================================================
extern "C" void kernel_launch(void* const* d_in, const int* in_sizes, int n_in,
                              void* d_out, int out_size, void* d_ws, size_t ws_size,
                              hipStream_t stream) {
  (void)in_sizes; (void)n_in; (void)out_size; (void)ws_size;
  const float* feat   = (const float*)d_in[0];
  const float* ln1g   = (const float*)d_in[1];
  const float* ln1b   = (const float*)d_in[2];
  const float* proj_w = (const float*)d_in[3];
  const float* proj_b = (const float*)d_in[4];
  const float* ln2g   = (const float*)d_in[5];
  const float* ln2b   = (const float*)d_in[6];
  const float* frame  = (const float*)d_in[7];
  const float* slot_e = (const float*)d_in[8];
  const float* nig    = (const float*)d_in[9];
  const float* nib    = (const float*)d_in[10];
  const float* nsg    = (const float*)d_in[11];
  const float* nsb    = (const float*)d_in[12];
  const float* wk     = (const float*)d_in[13];
  const float* wv     = (const float*)d_in[14];
  const float* wq     = (const float*)d_in[15];
  const float* p_scale= (const float*)d_in[16];
  const float* p_sink = (const float*)d_in[17];
  const float* p_gate = (const float*)d_in[18];
  const float* wih    = (const float*)d_in[19];
  const float* whh    = (const float*)d_in[20];
  const float* bih    = (const float*)d_in[21];
  const float* bhh    = (const float*)d_in[22];
  const float* nmg    = (const float*)d_in[23];
  const float* nmb    = (const float*)d_in[24];
  const float* mlp_w1 = (const float*)d_in[25];
  const float* mlp_b1 = (const float*)d_in[26];
  const float* mlp_w2 = (const float*)d_in[27];
  const float* mlp_b2 = (const float*)d_in[28];
  const float* pos    = (const float*)d_in[29];

  float* out_f = (float*)d_out;
  float* slots = out_f;                                   // [4,33,256]
  const size_t ATT0 = 4 * KSLOT * DN;                     // 33792
  float* attn_out   = out_f + ATT0;                       // [4,33,10952]
  float* attns_base = out_f + ATT0 + (size_t)4 * KSLOT * NPB;   // [3,4,33,10952]

  // ---- workspace carve-up (256B aligned) ----
  char* w = (char*)d_ws; size_t off = 0;
  auto take = [&](size_t bytes) -> void* {
    void* p = w + off; off += (bytes + 255) & ~(size_t)255; return p;
  };
  float2* stats  = (float2*)take((size_t)NTOK * 8);
  h16*  WPt      = (h16*)take((size_t)256 * 2048 * 2);
  h16*  inp      = (h16*)take((size_t)NTOK * 256 * 2);
  h16*  Wkvt     = (h16*)take((size_t)512 * 256 * 2);
  h16*  k_h      = (h16*)take((size_t)NTOK * 256 * 2);
  h16*  vT       = (h16*)take((size_t)4 * 256 * PADN * 2);
  h16*  q_h      = (h16*)take((size_t)4 * 48 * 256 * 2);
  float* wqT     = (float*)take((size_t)256 * 256 * 4);
  float* w1T     = (float*)take((size_t)512 * 256 * 4);
  float* w2T     = (float*)take((size_t)256 * 512 * 4);
  float* prevB   = (float*)take((size_t)4 * KSLOT * NPB * 4);
  float* logmean = (float*)take(1024);
  float* rowsum  = (float*)take(1024);
  h16*  attn_h   = (h16*)take((size_t)4 * 48 * PADN * 2);
  float* updates = (float*)take((size_t)4 * KSLOT * 256 * 4);

  // ---- prep ----
  k_transpose_f16<<<(2048 * 256 + 255) / 256, 256, 0, stream>>>(WPt, proj_w, 2048, 256);
  k_transpose_f16<<<(256 * 256 + 255) / 256, 256, 0, stream>>>(Wkvt, wk, 256, 256);
  k_transpose_f16<<<(256 * 256 + 255) / 256, 256, 0, stream>>>(Wkvt + 256 * 256, wv, 256, 256);
  k_transpose_f32<<<(256 * 256 + 255) / 256, 256, 0, stream>>>(wqT, wq, 256, 256);
  k_transpose_f32<<<(256 * 512 + 255) / 256, 256, 0, stream>>>(w1T, mlp_w1, 256, 512);
  k_transpose_f32<<<(512 * 256 + 255) / 256, 256, 0, stream>>>(w2T, mlp_w2, 512, 256);
  k_zero_vt_tail<<<(4 * 256 * (PADN - NPB) + 255) / 256, 256, 0, stream>>>(vT);
  k_init_slots<<<(4 * KSLOT * DN + 255) / 256, 256, 0, stream>>>(slots, slot_e);

  // ---- encoder ----
  k_ln_stats<<<NTOK, 256, 0, stream>>>(feat, stats);
  k_proj_gemm<<<NTOK / 32, 256, 0, stream>>>(feat, stats, ln1g, ln1b, WPt, proj_b,
                                             ln2g, ln2b, nig, nib, pos, frame, inp);
  k_kv_gemm<<<NTOK / 32, 256, 0, stream>>>(inp, Wkvt, k_h, vT);

  // ---- slot attention iterations ----
  for (int t = 0; t < 3; ++t){
    k_slots_q<<<4 * 48, 256, 0, stream>>>(slots, nsg, nsb, wqT, q_h);
    k_attn<<<dim3(PADN / 128, 4), 256, 0, stream>>>(q_h, k_h, attns_base, attn_h,
                                                    prevB, logmean,
                                                    p_scale, p_sink, p_gate, t);
    k_attn_reduce<<<4 * KSLOT, 256, 0, stream>>>(attns_base + (size_t)t * 4 * KSLOT * NPB,
                                                 rowsum, logmean);
    k_updates<<<dim3(2, 4), 256, 0, stream>>>(attn_h, vT, rowsum, updates);
    k_gru_mlp<<<4 * KSLOT, 256, 0, stream>>>(updates, slots, wih, whh, bih, bhh,
                                             nmg, nmb, w1T, mlp_b1, w2T, mlp_b2);
  }

  // attn output = all_attns[2]
  hipMemcpyAsync(attn_out, attns_base + (size_t)2 * 4 * KSLOT * NPB,
                 (size_t)4 * KSLOT * NPB * sizeof(float),
                 hipMemcpyDeviceToDevice, stream);
}
